// Attention_42502996361360
// MI455X (gfx1250) — matrix operand order repaired and checked by reading, not compiled, not run on hardware
//
#include <hip/hip_runtime.h>

// Problem geometry (fixed by the reference)
#define B_    16
#define N1_   512
#define N2_   512
#define F_    1024     // F1 == F2
#define HID_  512
#define DEG_  32
#define ROWS_ (B_ * N1_)        // 8192 rows per tensor
#define NNZ_  (B_ * N1_ * DEG_) // 262144

typedef __attribute__((ext_vector_type(16))) __bf16          v16bf;
typedef __attribute__((ext_vector_type(16))) unsigned short  v16u;
typedef __attribute__((ext_vector_type(8)))  unsigned short  v8u;
typedef __attribute__((ext_vector_type(8)))  float           v8f;
typedef __attribute__((ext_vector_type(4)))  float           v4f;

// f32 -> bf16 with round-to-nearest-even (bit-exact RNE)
__device__ __forceinline__ unsigned short f32_bf16_rne(float f) {
    unsigned int u = __builtin_bit_cast(unsigned int, f);
    unsigned int r = 0x7FFFu + ((u >> 16) & 1u);
    return (unsigned short)((u + r) >> 16);
}

// ---------------------------------------------------------------------------
// Kernel 1: fold v through the linear layers: u{1,2}[f] = sum_h W{1,2}[h,f]*v[h]
// (b1.v / b2.v are global constants -> cancel in segment softmax, dropped)
// Output: u_bf[0..1023] = bf16(u1), u_bf[1024..2047] = bf16(u2)
// ---------------------------------------------------------------------------
__global__ void __launch_bounds__(256)
fold_uv_kernel(const float* __restrict__ W1, const float* __restrict__ W2,
               const float* __restrict__ v, unsigned short* __restrict__ u_bf) {
    __shared__ float vs[HID_];
    for (int h = threadIdx.x; h < HID_; h += blockDim.x) vs[h] = v[h];
    __syncthreads();

    int gid = blockIdx.x * blockDim.x + threadIdx.x;   // 0..2047
    const float* W = (gid < F_) ? W1 : W2;
    int f = gid & (F_ - 1);
    float acc = 0.f;
#pragma unroll 8
    for (int h = 0; h < HID_; ++h)                     // coalesced: lane ~ f
        acc = fmaf(W[h * F_ + f], vs[h], acc);
    u_bf[gid] = f32_bf16_rne(acc);
}

// ---------------------------------------------------------------------------
// Kernel 2: p[mat][row] = dot(T[row, 0:1024], u[mat])  via WMMA bf16.
// One wave handles a 16-row tile. B = u chunk broadcast to all 16 columns
// (every C column holds the same partial sums); C accumulates f32 over K.
//
// A (16x32 bf16) per ISA layout:  M = lane&15, half = lane>>4,
//   a[0..7]  = K = half*8 + {0..7},  a[8..15] = K = 16 + half*8 + {0..7}
// B (32x16 bf16): lanes0-15 hold K=0..7,16..23; lanes16-31 K=8..15,24..31 (as A)
//   -> lane needs u[kc + half*8 + {0..7}] and u[kc + 16 + half*8 + {0..7}]
// C (16x16 f32): lane 0 VGPR r = row r; lane 16 VGPR r = row 8+r
// ---------------------------------------------------------------------------
__global__ void __launch_bounds__(256)
gemv_wmma_kernel(const float* __restrict__ t1, const float* __restrict__ t2,
                 const unsigned short* __restrict__ u_bf,
                 float* __restrict__ p /* [2][ROWS_] */) {
    int wid  = (blockIdx.x * blockDim.x + threadIdx.x) >> 5; // 0..1023
    int lane = threadIdx.x & 31;
    int mat      = wid >> 9;            // 512 waves per matrix
    int row_base = (wid & 511) << 4;    // 16 rows per wave

    const float*          T = mat ? t2 : t1;
    const unsigned short* U = u_bf + mat * F_;
    float*                P = p + mat * ROWS_;

    int m    = lane & 15;
    int half = lane >> 4;
    const float* trow = T + (size_t)(row_base + m) * F_;

    v8f c = {};
    for (int kc = 0; kc < F_; kc += 32) {
        const float* ap = trow + kc + half * 8;
        v4f f0 = *(const v4f*)(ap);
        v4f f1 = *(const v4f*)(ap + 4);
        v4f f2 = *(const v4f*)(ap + 16);
        v4f f3 = *(const v4f*)(ap + 20);
        if (kc + 32 < F_) __builtin_prefetch(ap + 32, 0, 1); // global_prefetch_b8

        v16u au;
#pragma unroll
        for (int j = 0; j < 4; ++j) {
            au[j]      = f32_bf16_rne(f0[j]);
            au[j + 4]  = f32_bf16_rne(f1[j]);
            au[j + 8]  = f32_bf16_rne(f2[j]);
            au[j + 12] = f32_bf16_rne(f3[j]);
        }

        const unsigned short* up = U + kc + half * 8;
        v8u b0 = *(const v8u*)(up);
        v8u b1 = *(const v8u*)(up + 16);
        v16u bu;
#pragma unroll
        for (int j = 0; j < 8; ++j) { bu[j] = b0[j]; bu[j + 8] = b1[j]; }

        v16bf a = __builtin_bit_cast(v16bf, au);
        v16bf b = __builtin_bit_cast(v16bf, bu);
        // (neg_a, A, neg_b, B, c_mod, C, reuse_a, reuse_b)
        c = __builtin_amdgcn_wmma_f32_16x16x32_bf16(false, a, false, b,
                                                    (short)0, c, false, false);
    }

    // All columns identical; lane 0 stores rows 0..7, lane 16 rows 8..15.
    if (m == 0) {
        float* dst = P + row_base + half * 8;
        v4f lo = { c[0], c[1], c[2], c[3] };
        v4f hi = { c[4], c[5], c[6], c[7] };
        *(v4f*)dst       = lo;
        *(v4f*)(dst + 4) = hi;
    }
}

// ---------------------------------------------------------------------------
// Kernel 3: w[n] = p1[b*N1+i] + p2[b*N2+j]; softmax over each contiguous
// 32-entry (b,i) segment. One wave32 per segment — shuffle reductions only.
// ---------------------------------------------------------------------------
__global__ void __launch_bounds__(256)
segsoftmax_kernel(const int* __restrict__ idx_b, const int* __restrict__ idx_i,
                  const int* __restrict__ idx_j,
                  const float* __restrict__ p /* [2][ROWS_] */,
                  float* __restrict__ out) {
    int n = blockIdx.x * blockDim.x + threadIdx.x;     // nnz index
    int b = idx_b[n], i = idx_i[n], j = idx_j[n];
    float w = p[b * N1_ + i] + p[ROWS_ + b * N2_ + j];

    float mx = w;
#pragma unroll
    for (int off = 16; off > 0; off >>= 1)
        mx = fmaxf(mx, __shfl_xor(mx, off, 32));
    float e = __expf(w - mx);
    float s = e;
#pragma unroll
    for (int off = 16; off > 0; off >>= 1)
        s += __shfl_xor(s, off, 32);
    out[n] = e / s;
}

// ---------------------------------------------------------------------------
extern "C" void kernel_launch(void* const* d_in, const int* in_sizes, int n_in,
                              void* d_out, int out_size, void* d_ws, size_t ws_size,
                              hipStream_t stream) {
    const float* t1    = (const float*)d_in[0];
    const float* t2    = (const float*)d_in[1];
    const int*   idx_b = (const int*)d_in[2];
    const int*   idx_i = (const int*)d_in[3];
    const int*   idx_j = (const int*)d_in[4];
    const float* W1    = (const float*)d_in[5];
    // d_in[6] = b1, d_in[8] = b2: constants through v-dot, cancel in softmax
    const float* W2    = (const float*)d_in[7];
    const float* v     = (const float*)d_in[9];

    // Workspace: [u_bf16: 2*1024*2B = 4KB][p: 2*8192 f32 = 64KB]
    unsigned short* u_bf = (unsigned short*)d_ws;
    float*          p    = (float*)((char*)d_ws + 2 * F_ * sizeof(unsigned short));

    fold_uv_kernel<<<(2 * F_) / 256, 256, 0, stream>>>(W1, W2, v, u_bf);
    gemv_wmma_kernel<<<(2 * ROWS_ / 16) * 32 / 256, 256, 0, stream>>>(t1, t2, u_bf, p);
    segsoftmax_kernel<<<NNZ_ / 256, 256, 0, stream>>>(idx_b, idx_i, idx_j, p,
                                                      (float*)d_out);
}
